// GIUNetSpect_44985487458607
// MI455X (gfx1250) — matrix operand order repaired and checked by reading, not compiled, not run on hardware
//
#include <hip/hip_runtime.h>
#include <hip/hip_bf16.h>
#include <cstdint>

// ---------------------------------------------------------------------------
// Graph U-Net (GIUNet) forward for MI455X (gfx1250, wave32).
// Dense linear stages: v_wmma_f32_16x16x32_f16, fragments via ds_load_b128,
// weights pre-converted to f16 and staged into LDS with
// global_load_async_to_lds_b128 (ASYNCcnt) when available.
// Edge aggregation / pooling / unpooling are memory-bound scatter kernels.
// ---------------------------------------------------------------------------

typedef __attribute__((ext_vector_type(16))) _Float16 v16h;
typedef __attribute__((ext_vector_type(8)))  _Float16 v8h;
typedef __attribute__((ext_vector_type(4)))  _Float16 v4h;
typedef __attribute__((ext_vector_type(8)))  float    v8f;
typedef int v4i __attribute__((vector_size(16)));   // builtin's b128 element type

#if defined(__gfx1250__) && \
    __has_builtin(__builtin_amdgcn_global_load_async_to_lds_b128) && \
    __has_builtin(__builtin_amdgcn_s_wait_asynccnt)
#define USE_ASYNC_LDS 1
typedef __attribute__((address_space(1))) v4i* gv4i_p;
typedef __attribute__((address_space(3))) v4i* lv4i_p;
#else
#define USE_ASYNC_LDS 0
#endif

static constexpr int NN  = 100000;   // nodes
static constexpr int EE  = 1600000;  // edges
static constexpr int K1C = 80000;    // pool1 keep
static constexpr int K2C = 64000;    // pool2 keep
static constexpr int NG  = 64;       // graphs
static constexpr int NC  = 10;       // classes
static constexpr int NP  = 1 << 17;  // bitonic sort size (>= NN)

// ------------------------------- utility kernels ---------------------------

__global__ void zero_f(float* p, int n) {
    int i = blockIdx.x * blockDim.x + threadIdx.x;
    if (i < n) p[i] = 0.f;
}

__global__ void zero_f4(float* p, int n4) {   // n4 = count of float4
    int i = blockIdx.x * blockDim.x + threadIdx.x;
    if (i < n4) *(float4*)(p + 4 * (long)i) = make_float4(0.f, 0.f, 0.f, 0.f);
}

__global__ void fill_neg1(int* p, int n) {
    int i = blockIdx.x * blockDim.x + threadIdx.x;
    if (i < n) p[i] = -1;
}

__global__ void add_inplace4(float* __restrict__ a, const float* __restrict__ b, int n4) {
    int i = blockIdx.x * blockDim.x + threadIdx.x;
    if (i >= n4) return;
    float4* pa = (float4*)(a + 4 * (long)i);
    const float4 vb = *(const float4*)(b + 4 * (long)i);
    float4 va = *pa;
    va.x += vb.x; va.y += vb.y; va.z += vb.z; va.w += vb.w;
    *pa = va;
}

__global__ void relu_k(float* p, int n) {
    int i = blockIdx.x * blockDim.x + threadIdx.x;
    if (i < n) { float v = p[i]; p[i] = v > 0.f ? v : 0.f; }
}

// Convert W[din,dout] (f32, row-major) -> Wt[doutp, din] (f16, transposed, padded)
__global__ void wcvt(const float* __restrict__ W, _Float16* __restrict__ Wt,
                     int din, int dout, int doutp) {
    int i = blockIdx.x * blockDim.x + threadIdx.x;
    if (i >= doutp * din) return;
    int c = i / din, k = i - c * din;
    Wt[i] = (_Float16)(c < dout ? W[k * dout + c] : 0.f);
}

// --------------------------- edge aggregation ------------------------------
// One thread per (edge, 4 features): vector gather + 4 float atomics.

__global__ void edge_agg4(const float* __restrict__ x,
                          const int* __restrict__ src,
                          const int* __restrict__ dst,
                          const float* __restrict__ m,
                          float* __restrict__ agg, int e, int f) {
    int q = f >> 2;                       // feature groups of 4
    int i = blockIdx.x * blockDim.x + threadIdx.x;
    if (i >= e * q) return;
    int ed = i / q;
    int c  = (i - ed * q) << 2;
    float mm = m ? m[ed] : 1.f;
    if (mm == 0.f) return;
    const float4 v = *(const float4*)(x + (long)src[ed] * f + c);
    float* a = agg + (long)dst[ed] * f + c;
    atomicAdd(a + 0, v.x * mm);
    atomicAdd(a + 1, v.y * mm);
    atomicAdd(a + 2, v.z * mm);
    atomicAdd(a + 3, v.w * mm);
}

// ------------------------------ WMMA GEMM ----------------------------------
// Y[n, DOUT] = X[n, DIN] @ W[DIN, DOUT] + bias      (f16 inputs, f32 accum)
// 256 threads = 8 waves; each wave owns one 16-row stripe, loops col tiles.
// Wt: pre-transposed f16 weights [DOUTP, DIN] -> B fragment is one 32B run.

template <int DIN, int DOUT>
__global__ void __launch_bounds__(256)
gemm_bias(const float* __restrict__ X, const _Float16* __restrict__ Wt,
          const float* __restrict__ bias, float* __restrict__ Y, int n) {
    constexpr int DOUTP = (DOUT + 15) & ~15;
    __shared__ _Float16 sX[128 * DIN];
    __shared__ _Float16 sWt[DOUTP * DIN];   // transposed: sWt[col*DIN + k]
    __shared__ float    sB[DOUTP];

    const int tid      = threadIdx.x;
    const int rowBase  = blockIdx.x * 128;
    const bool fullBlk = (rowBase + 128 <= n);

    // stage weights: straight byte copy -> async DMA to LDS (ASYNCcnt path)
#if USE_ASYNC_LDS
    for (int i8 = tid; i8 < (DOUTP * DIN) / 8; i8 += 256) {
        __builtin_amdgcn_global_load_async_to_lds_b128(
            (gv4i_p)(Wt + i8 * 8), (lv4i_p)&sWt[i8 * 8], 0, 0);
    }
#else
    for (int i8 = tid; i8 < (DOUTP * DIN) / 8; i8 += 256)
        *(v8h*)&sWt[i8 * 8] = *(const v8h*)&Wt[i8 * 8];
#endif
    if (tid < DOUTP) sB[tid] = (tid < DOUT) ? bias[tid] : 0.f;

    // stage X tile (float4 gather -> packed v4h store) — overlaps weight DMA
    for (int i4 = tid; i4 < (128 * DIN) / 4; i4 += 256) {
        int i = i4 * 4;
        int r = i / DIN, c = i - r * DIN;
        int gr = rowBase + r;
        float4 v;
        if (fullBlk || gr < n) v = *(const float4*)(X + (long)gr * DIN + c);
        else                   v = make_float4(0.f, 0.f, 0.f, 0.f);
        v4h h4 = { (_Float16)v.x, (_Float16)v.y, (_Float16)v.z, (_Float16)v.w };
        *(v4h*)&sX[i] = h4;
    }
#if USE_ASYNC_LDS
    __builtin_amdgcn_s_wait_asynccnt(0);
#endif
    __syncthreads();

    const int wave = tid >> 5;
    const int lane = tid & 31;
    const int half = lane >> 4;   // 0: lanes 0-15, 1: lanes 16-31
    const int ln   = lane & 15;
    const int arow = wave * 16 + ln;   // A-matrix row held by this lane

    for (int ct = 0; ct < DOUTP / 16; ++ct) {
        const int col = ct * 16 + ln;
        v8f acc = {};
#pragma unroll
        for (int kt = 0; kt < DIN / 32; ++kt) {
            // A 16x32 f16: frag = K[kb..kb+15], contiguous (matches B)
            const int kb = kt * 32 + (half ? 16 : 0);
            v8h alo = *(const v8h*)&sX[arow * DIN + kb];
            v8h ahi = *(const v8h*)&sX[arow * DIN + kb + 8];
            v16h a = __builtin_shufflevector(alo, ahi,
                       0,1,2,3,4,5,6,7,8,9,10,11,12,13,14,15);
            // B 32x16 f16: frag = K[kc..kc+15], contiguous in transposed sWt
            const int kc = kt * 32 + (half ? 16 : 0);
            v8h blo = *(const v8h*)&sWt[col * DIN + kc];
            v8h bhi = *(const v8h*)&sWt[col * DIN + kc + 8];
            v16h b = __builtin_shufflevector(blo, bhi,
                       0,1,2,3,4,5,6,7,8,9,10,11,12,13,14,15);
            acc = __builtin_amdgcn_wmma_f32_16x16x32_f16(
                false, a, false, b, (short)0, acc, false, false);
        }
        const float bb   = sB[col];
        const bool colOk = (DOUT == DOUTP) || (col < DOUT);
        const int  gr0   = rowBase + wave * 16 + (half ? 8 : 0);  // C/D: M = v + 8*half
        if (fullBlk && colOk) {
#pragma unroll
            for (int v = 0; v < 8; ++v)
                Y[(long)(gr0 + v) * DOUT + col] = acc[v] + bb;
        } else if (colOk) {
#pragma unroll
            for (int v = 0; v < 8; ++v)
                if (gr0 + v < n) Y[(long)(gr0 + v) * DOUT + col] = acc[v] + bb;
        }
    }
}

// ------------------------------- batchnorm ---------------------------------

__global__ void bn_stats(const float* __restrict__ Y, int n, int d,
                         float* __restrict__ stats) {  // [0..d)=mean, [d..2d)=rsqrt(var+eps)
    __shared__ float sh[256], sh2[256];
    int col = blockIdx.x;
    float s = 0.f, s2 = 0.f;
    for (int r = threadIdx.x; r < n; r += 256) {
        float v = Y[(long)r * d + col];
        s += v; s2 += v * v;
    }
    sh[threadIdx.x] = s; sh2[threadIdx.x] = s2;
    __syncthreads();
    for (int off = 128; off > 0; off >>= 1) {
        if (threadIdx.x < off) {
            sh[threadIdx.x]  += sh[threadIdx.x + off];
            sh2[threadIdx.x] += sh2[threadIdx.x + off];
        }
        __syncthreads();
    }
    if (threadIdx.x == 0) {
        float mu  = sh[0] / n;
        float var = sh2[0] / n - mu * mu;
        stats[col]     = mu;
        stats[d + col] = rsqrtf(var + 1e-5f);
    }
}

__global__ void bn_apply_relu(float* __restrict__ Y, int n, int d,
                              const float* __restrict__ g,
                              const float* __restrict__ beta,
                              const float* __restrict__ stats) {
    int i = blockIdx.x * blockDim.x + threadIdx.x;
    if (i >= n * d) return;
    int c = i % d;
    float v = (Y[i] - stats[c]) * stats[d + c] * g[c] + beta[c];
    Y[i] = v > 0.f ? v : 0.f;
}

// ------------------------------ pooling ------------------------------------

__global__ void pool_scores(const float* __restrict__ h, const float* __restrict__ C,
                            const float* __restrict__ Wf, const float* __restrict__ bf,
                            const float* __restrict__ Ws, const float* __restrict__ bs,
                            const float* __restrict__ Wfin, const float* __restrict__ bfin,
                            float* __restrict__ out, int n, int f) {
    int i = blockIdx.x * blockDim.x + threadIdx.x;
    if (i >= n) return;
    float fw = bf[0];
    for (int c = 0; c < f; ++c) fw += h[(long)i * f + c] * Wf[c];
    float sw = bs[0];
    for (int c = 0; c < 4; ++c) sw += C[i * 4 + c] * Ws[c];
    float z = Wfin[0] * fw + Wfin[1] * sw + bfin[0];
    out[i] = 1.f / (1.f + expf(-z));
}

// descending-score sort key: ascending sort of (~bits(score), idx)
__global__ void pack_keys(const float* __restrict__ sc, unsigned long long* __restrict__ kb,
                          int n, int np) {
    int i = blockIdx.x * blockDim.x + threadIdx.x;
    if (i >= np) return;
    if (i < n) {
        unsigned int b = __float_as_uint(sc[i]);   // sigmoid > 0 -> bits monotone
        kb[i] = ((unsigned long long)(~b) << 32) | (unsigned int)i;
    } else {
        kb[i] = 0xFFFFFFFFFFFFFFFFull;
    }
}

__global__ void bitonic_step(unsigned long long* __restrict__ d, int j, int k) {
    int i   = blockIdx.x * blockDim.x + threadIdx.x;
    int ixj = i ^ j;
    if (ixj > i) {
        unsigned long long a = d[i], b = d[ixj];
        bool asc = ((i & k) == 0);
        if ((asc && a > b) || (!asc && a < b)) { d[i] = b; d[ixj] = a; }
    }
}

__global__ void extract_topk(const unsigned long long* __restrict__ kb,
                             const float* __restrict__ sc,
                             int* __restrict__ idx, float* __restrict__ vals, int k) {
    int i = blockIdx.x * blockDim.x + threadIdx.x;
    if (i >= k) return;
    int id = (int)(kb[i] & 0xFFFFFFFFu);
    idx[i]  = id;
    vals[i] = sc[id];
}

__global__ void gather_scale4(const float* __restrict__ h, const int* __restrict__ idx,
                              const float* __restrict__ vals, float* __restrict__ out,
                              int k, int f) {
    int q = f >> 2;
    int i = blockIdx.x * blockDim.x + threadIdx.x;
    if (i >= k * q) return;
    int r = i / q, c = (i - r * q) << 2;
    float s = vals[r];
    float4 v = *(const float4*)(h + (long)idx[r] * f + c);
    v.x *= s; v.y *= s; v.z *= s; v.w *= s;
    *(float4*)(out + (long)r * f + c) = v;
}

__global__ void scatter_pos(int* __restrict__ pos, const int* __restrict__ idx, int k) {
    int i = blockIdx.x * blockDim.x + threadIdx.x;
    if (i < k) pos[idx[i]] = i;
}

__global__ void remap_edges(const int* __restrict__ sIn, const int* __restrict__ dIn,
                            const float* __restrict__ mIn, const int* __restrict__ pos,
                            int* __restrict__ sOut, int* __restrict__ dOut,
                            float* __restrict__ mOut, int e) {
    int i = blockIdx.x * blockDim.x + threadIdx.x;
    if (i >= e) return;
    int ns = pos[sIn[i]];
    int nd = pos[dIn[i]];
    float mm = (mIn ? mIn[i] : 1.f) * ((ns >= 0) ? 1.f : 0.f) * ((nd >= 0) ? 1.f : 0.f);
    sOut[i] = ns < 0 ? 0 : ns;
    dOut[i] = nd < 0 ? 0 : nd;
    mOut[i] = mm;
}

__global__ void scatter_rows4(const float* __restrict__ in, const int* __restrict__ idx,
                              float* __restrict__ out, int k, int f) {
    int q = f >> 2;
    int i = blockIdx.x * blockDim.x + threadIdx.x;
    if (i >= k * q) return;
    int r = i / q, c = (i - r * q) << 2;
    *(float4*)(out + (long)idx[r] * f + c) = *(const float4*)(in + (long)r * f + c);
}

// ----------------------------- readout -------------------------------------

__global__ void seg_accum(const float* __restrict__ x, const int* __restrict__ batch,
                          float* __restrict__ sums, float* __restrict__ cnts, int n, int f) {
    int i = blockIdx.x * blockDim.x + threadIdx.x;
    if (i >= n) return;
    int g = batch[i];
    for (int c = 0; c < f; ++c) atomicAdd(&sums[g * f + c], x[(long)i * f + c]);
    atomicAdd(&cnts[g], 1.f);
}

__global__ void seg_finalize(const float* __restrict__ sums, const float* __restrict__ cnts,
                             float* __restrict__ out, int total, int f) {
    int i = blockIdx.x * blockDim.x + threadIdx.x;
    if (i >= total) return;
    int g = i / f;
    out[i] = sums[i] / fmaxf(cnts[g], 1.f);
}

// --------------------------- host-side helpers -----------------------------

static inline int cdiv(int a, int b) { return (a + b - 1) / b; }

static void run_agg_add(const float* xin, const int* sIn, const int* dIn, const float* mIn,
                        float* t0, int n, int f, hipStream_t st) {
    int tot4 = (n * f) >> 2;
    zero_f4<<<cdiv(tot4, 256), 256, 0, st>>>(t0, tot4);
    int etot = EE * (f >> 2);
    edge_agg4<<<cdiv(etot, 256), 256, 0, st>>>(xin, sIn, dIn, mIn, t0, EE, f);
    add_inplace4<<<cdiv(tot4, 256), 256, 0, st>>>(t0, xin, tot4);
}

template <int DIN, int DOUT>
static void run_linear_bn_relu(const float* X, const _Float16* Wt, const float* b,
                               const float* g, const float* beta,
                               float* Y, float* stats, int n, hipStream_t st) {
    gemm_bias<DIN, DOUT><<<cdiv(n, 128), 256, 0, st>>>(X, Wt, b, Y, n);
    bn_stats<<<DOUT, 256, 0, st>>>(Y, n, DOUT, stats);
    int tot = n * DOUT;
    bn_apply_relu<<<cdiv(tot, 256), 256, 0, st>>>(Y, n, DOUT, g, beta, stats);
}

static void run_pool(const float* h, const float* Cc,
                     const float* Wf, const float* bf, const float* Ws, const float* bs,
                     const float* Wfin, const float* bfin,
                     const int* sIn, const int* dIn, const float* mIn,
                     float* newh, int* idx, float* vals, int* pos,
                     float* scores, unsigned long long* sortb,
                     int* sOut, int* dOut, float* mOut,
                     int n, int k, int f, hipStream_t st) {
    pool_scores<<<cdiv(n, 256), 256, 0, st>>>(h, Cc, Wf, bf, Ws, bs, Wfin, bfin, scores, n, f);
    pack_keys<<<NP / 256, 256, 0, st>>>(scores, sortb, n, NP);
    for (int kk = 2; kk <= NP; kk <<= 1)
        for (int jj = kk >> 1; jj > 0; jj >>= 1)
            bitonic_step<<<NP / 256, 256, 0, st>>>(sortb, jj, kk);
    extract_topk<<<cdiv(k, 256), 256, 0, st>>>(sortb, scores, idx, vals, k);
    gather_scale4<<<cdiv(k * (f >> 2), 256), 256, 0, st>>>(h, idx, vals, newh, k, f);
    fill_neg1<<<cdiv(n, 256), 256, 0, st>>>(pos, n);
    scatter_pos<<<cdiv(k, 256), 256, 0, st>>>(pos, idx, k);
    remap_edges<<<cdiv(EE, 256), 256, 0, st>>>(sIn, dIn, mIn, pos, sOut, dOut, mOut, EE);
}

// ------------------------------ entry point --------------------------------

extern "C" void kernel_launch(void* const* d_in, const int* in_sizes, int n_in,
                              void* d_out, int out_size, void* d_ws, size_t ws_size,
                              hipStream_t stream) {
    (void)in_sizes; (void)n_in; (void)out_size; (void)ws_size;

    // ---- inputs, setup_inputs() insertion order (params nested in order) ----
    const float* x     = (const float*)d_in[0];
    const int*   eidx  = (const int*)d_in[1];
    const int*   batch = (const int*)d_in[2];
    const float* C1    = (const float*)d_in[3];
    const float* C2    = (const float*)d_in[4];
    auto F = [&](int i) { return (const float*)d_in[i]; };
    // conv1: 5..12 | pool1: 13..18 | conv2: 19..26 | pool2: 27..32
    // midconv: 33..40 | decoder2: 41..48 | decoder1: 49..50
    const int* src0 = eidx;
    const int* dst0 = eidx + EE;

    // ---- workspace bump allocator ----
    char* wp = (char*)d_ws;
    auto alloc = [&](size_t bytes) -> char* {
        char* r = wp;
        wp += (bytes + 255) & ~(size_t)255;
        return r;
    };
    float* big1 = (float*)alloc((size_t)NN * 64 * sizeof(float));
    float* big2 = (float*)alloc((size_t)NN * 64 * sizeof(float));
    float* big3 = (float*)alloc((size_t)NN * 64 * sizeof(float));
    int*   s1   = (int*)alloc((size_t)EE * 4);
    int*   d1   = (int*)alloc((size_t)EE * 4);
    float* m1   = (float*)alloc((size_t)EE * 4);
    int*   s2   = (int*)alloc((size_t)EE * 4);
    int*   d2   = (int*)alloc((size_t)EE * 4);
    float* m2   = (float*)alloc((size_t)EE * 4);
    float* scores = (float*)alloc((size_t)NN * 4);
    unsigned long long* sortb = (unsigned long long*)alloc((size_t)NP * 8);
    int*   idx1  = (int*)alloc((size_t)K1C * 4);
    float* vals1 = (float*)alloc((size_t)K1C * 4);
    int*   idx2  = (int*)alloc((size_t)K2C * 4);
    float* vals2 = (float*)alloc((size_t)K2C * 4);
    int*   pos   = (int*)alloc((size_t)NN * 4);
    float* stats = (float*)alloc(256 * 4);
    float* sums  = (float*)alloc((size_t)NG * NC * 4);
    float* cnts  = (float*)alloc((size_t)NG * 4);
    _Float16* wtArena = (_Float16*)alloc((size_t)32768 * 2);

    // ---- pre-convert all weights: f32 [din,dout] -> f16 [doutp,din] ----
    _Float16* wt = wtArena;
    auto prep_w = [&](const float* W, int din, int dout) -> _Float16* {
        int doutp = (dout + 15) & ~15;
        int tot = doutp * din;
        _Float16* dst = wt;
        wcvt<<<cdiv(tot, 256), 256, 0, stream>>>(W, dst, din, dout, doutp);
        wt += (tot + 127) & ~127;
        return dst;
    };
    _Float16* Wc1a = prep_w(F(5),  64, 32);
    _Float16* Wc1b = prep_w(F(9),  32, 32);
    _Float16* Wc2a = prep_w(F(19), 32, 64);
    _Float16* Wc2b = prep_w(F(23), 64, 64);
    _Float16* Wma  = prep_w(F(33), 64, 64);
    _Float16* Wmb  = prep_w(F(37), 64, 64);
    _Float16* Wd2a = prep_w(F(41), 64, 32);
    _Float16* Wd2b = prep_w(F(45), 32, 32);
    _Float16* Wd1  = prep_w(F(49), 32, 10);

    // ---- conv1: GIN(64 -> 32) on N nodes, full edge set ----
    run_agg_add(x, src0, dst0, nullptr, big1, NN, 64, stream);
    run_linear_bn_relu<64, 32>(big1, Wc1a, F(6), F(7), F(8), big2, stats, NN, stream);
    run_linear_bn_relu<32, 32>(big2, Wc1b, F(10), F(11), F(12), big3, stats, NN, stream);
    // x1 = big3 [N,32] (already >= 0; outer relu is a no-op)

    // ---- pool1: N -> K1, h1 = big2 [K1,32], edges -> s1/d1/m1 ----
    run_pool(big3, C1, F(13), F(14), F(15), F(16), F(17), F(18),
             src0, dst0, nullptr,
             big2, idx1, vals1, pos, scores, sortb,
             s1, d1, m1, NN, K1C, 32, stream);

    // ---- conv2: GIN(32 -> 64) on K1 nodes ----
    run_agg_add(big2, s1, d1, m1, big1, K1C, 32, stream);
    run_linear_bn_relu<32, 64>(big1, Wc2a, F(20), F(21), F(22), big3, stats, K1C, stream);
    run_linear_bn_relu<64, 64>(big3, Wc2b, F(24), F(25), F(26), big2, stats, K1C, stream);
    // x2 = big2 [K1,64]

    // ---- pool2: K1 -> K2, h2 = big1 [K2,64], edges -> s2/d2/m2 ----
    run_pool(big2, C2, F(27), F(28), F(29), F(30), F(31), F(32),
             s1, d1, m1,
             big1, idx2, vals2, pos, scores, sortb,
             s2, d2, m2, K1C, K2C, 64, stream);

    // ---- midconv: GIN(64 -> 64) on K2 nodes ----
    run_agg_add(big1, s2, d2, m2, big3, K2C, 64, stream);
    run_linear_bn_relu<64, 64>(big3, Wma, F(34), F(35), F(36), big2, stats, K2C, stream);
    run_linear_bn_relu<64, 64>(big2, Wmb, F(38), F(39), F(40), big1, stats, K2C, stream);
    // xm = big1 [K2,64]

    // ---- unpool2: scatter K2 rows into K1 slots -> xd2 = big3 [K1,64] ----
    zero_f4<<<cdiv(K1C * 16, 256), 256, 0, stream>>>(big3, K1C * 16);
    scatter_rows4<<<cdiv(K2C * 16, 256), 256, 0, stream>>>(big1, idx2, big3, K2C, 64);

    // ---- decoder2: GIN(64 -> 32) on K1 nodes, edges s2/d2/m2 ----
    run_agg_add(big3, s2, d2, m2, big2, K1C, 64, stream);
    run_linear_bn_relu<64, 32>(big2, Wd2a, F(42), F(43), F(44), big1, stats, K1C, stream);
    run_linear_bn_relu<32, 32>(big1, Wd2b, F(46), F(47), F(48), big3, stats, K1C, stream);
    // xd2o = big3 [K1,32]

    // ---- unpool1: scatter K1 rows into N slots -> xd1 = big2 [N,32] ----
    zero_f4<<<cdiv(NN * 8, 256), 256, 0, stream>>>(big2, NN * 8);
    scatter_rows4<<<cdiv(K1C * 8, 256), 256, 0, stream>>>(big3, idx1, big2, K1C, 32);

    // ---- decoder1: relu(linear 32 -> 10) -> big1 [N,10] ----
    gemm_bias<32, 10><<<cdiv(NN, 128), 256, 0, stream>>>(big2, Wd1, F(50), big1, NN);
    relu_k<<<cdiv(NN * NC, 256), 256, 0, stream>>>(big1, NN * NC);

    // ---- per-graph mean over batch ----
    zero_f<<<cdiv(NG * NC, 256), 256, 0, stream>>>(sums, NG * NC);
    zero_f<<<1, 256, 0, stream>>>(cnts, NG);
    seg_accum<<<cdiv(NN, 256), 256, 0, stream>>>(big1, batch, sums, cnts, NN, NC);
    seg_finalize<<<cdiv(NG * NC, 256), 256, 0, stream>>>(sums, cnts, (float*)d_out, NG * NC, NC);
}
